// CARCAttentionProcessor_58969900974798
// MI455X (gfx1250) — matrix operand order repaired and checked by reading, not compiled, not run on hardware
//
#include <hip/hip_runtime.h>

// ---------------------------------------------------------------------------
// CARC attention processor for MI455X (gfx1250), bf16 WMMA pipeline, v3.
//   B=4, L=1024, D=1280, H=20, DH=64, ALPHA=0.48
// v3: software-pipelined GEMM k-loop (double-buffered fragments) and
// attention kv-loop (K frags pipelined across iterations, V frags issued
// ahead of the softmax VALU block) so WMMAs overlap in-flight VMEM.
// ---------------------------------------------------------------------------

typedef __attribute__((ext_vector_type(16))) __bf16 v16bf;
typedef __attribute__((ext_vector_type(8)))  float  v8f;

#define D_MODEL 1280
#define SEQ     1024
#define NHEADS  20
#define DH      64
#define BATCH   4
#define BH      (BATCH * NHEADS)     // 80
#define KVLEN   (2 * SEQ)            // 2048
#define MROWS   (BATCH * SEQ)        // 4096
#define ALPHA   0.48f
#define SM_SCALE 0.125f              // 1/sqrt(64)

static __device__ __forceinline__ v8f zero8() {
  v8f z = {0.f, 0.f, 0.f, 0.f, 0.f, 0.f, 0.f, 0.f};
  return z;
}

// A fragment (16x32 bf16, M x K), row-major source, per CDNA5 ISA 7.12.2.
static __device__ __forceinline__ v16bf load_a_frag(const __bf16* base, int ld,
                                                    int m0, int k0) {
  const int lane = threadIdx.x & 31;
  const int m  = m0 + (lane & 15);
  const int kb = k0 + ((lane >> 4) << 3);
  const __bf16* p = base + (size_t)m * ld + kb;
  union U { uint4 u[2]; v16bf v; } x;
  x.u[0] = *reinterpret_cast<const uint4*>(p);
  x.u[1] = *reinterpret_cast<const uint4*>(p + 16);
  return x.v;
}

// B fragment (32x16 bf16, K x N) from a K-contiguous (N-major) source.
static __device__ __forceinline__ v16bf load_bT_frag(const __bf16* src, int ld,
                                                     int n0, int k0) {
  const int lane = threadIdx.x & 31;
  const int n  = n0 + (lane & 15);
  const int kb = k0 + ((lane >> 4) << 3);
  const __bf16* p = src + (size_t)n * ld + kb;
  union U { uint4 u[2]; v16bf v; } x;
  x.u[0] = *reinterpret_cast<const uint4*>(p);
  x.u[1] = *reinterpret_cast<const uint4*>(p + 16);
  return x.v;
}

// ---------------------------------------------------------------------------
// Conversion kernels
// ---------------------------------------------------------------------------
__global__ void carc_cvt_bf16(const float* __restrict__ src,
                              __bf16* __restrict__ dst, int n) {
  int i = blockIdx.x * blockDim.x + threadIdx.x;
  if (i < n) dst[i] = (__bf16)src[i];
}

// W[k][n] fp32 -> Wt[n][k] bf16 (1280x1280)
__global__ void carc_cvt_w_t(const float* __restrict__ src,
                             __bf16* __restrict__ dst, int n) {
  int i = blockIdx.x * blockDim.x + threadIdx.x;
  if (i < n) {
    int k = i / D_MODEL, c = i - k * D_MODEL;
    dst[(size_t)c * D_MODEL + k] = (__bf16)src[i];
  }
}

// K_bg [80,1024,64] fp32 -> alpha*bf16 into back half of K_ext [80,2048,64]
__global__ void carc_cvt_bg_k(const float* __restrict__ src,
                              __bf16* __restrict__ ext, int n) {
  int i = blockIdx.x * blockDim.x + threadIdx.x;
  if (i < n) {
    int bh = i >> 16, rest = i & 0xFFFF;
    ext[((size_t)bh << 17) + (1 << 16) + rest] = (__bf16)(ALPHA * src[i]);
  }
}

// V_bg [80,1024,64] fp32 -> alpha*bf16 transposed into Vt [80,64,2048] back half
__global__ void carc_cvt_bg_vT(const float* __restrict__ src,
                               __bf16* __restrict__ vt, int n) {
  int i = blockIdx.x * blockDim.x + threadIdx.x;
  if (i < n) {
    int bh = i >> 16, rest = i & 0xFFFF;
    int j = rest >> 6, dh = rest & 63;
    vt[((size_t)bh * DH + dh) * KVLEN + SEQ + j] = (__bf16)(ALPHA * src[i]);
  }
}

// ---------------------------------------------------------------------------
// GEMM: C[4096,1280] = A[4096,1280] @ W (W supplied TRANSPOSED: Wt[n][k]).
// Block = 128 threads (4 waves), each wave 16 rows x 64 cols.
// Software-pipelined: iteration k issues k+1's loads before k's WMMAs.
// mode 0: -> Q [bh,1024,64]; mode 1: -> K_ext front; mode 2: -> Vt front;
// mode 3: -> fp32 out = acc + bias.
// ---------------------------------------------------------------------------
__global__ void carc_gemm_wmma(const __bf16* __restrict__ A,
                               const __bf16* __restrict__ Wt,
                               int mode, const float* __restrict__ bias,
                               void* __restrict__ out) {
  const int wave = threadIdx.x >> 5;
  const int lane = threadIdx.x & 31;
  const int n0blk = blockIdx.x * 64;
  const int m0 = blockIdx.y * 64 + wave * 16;

  v8f acc[4] = {zero8(), zero8(), zero8(), zero8()};

  // prologue: fragments for k0 = 0
  v16bf af = load_a_frag(A, D_MODEL, m0, 0);
  v16bf bf[4];
#pragma unroll
  for (int i = 0; i < 4; ++i) bf[i] = load_bT_frag(Wt, D_MODEL, n0blk + i * 16, 0);

  for (int k0 = 32; k0 < D_MODEL; k0 += 32) {
    __builtin_prefetch(A + (size_t)m0 * D_MODEL + k0 + 32, 0, 1);
    // issue next chunk's loads first ...
    v16bf afn = load_a_frag(A, D_MODEL, m0, k0);
    v16bf bfn[4];
#pragma unroll
    for (int i = 0; i < 4; ++i)
      bfn[i] = load_bT_frag(Wt, D_MODEL, n0blk + i * 16, k0);
    // ... then consume the current fragments
#pragma unroll
    for (int i = 0; i < 4; ++i)
      acc[i] = __builtin_amdgcn_wmma_f32_16x16x32_bf16(
          false, af, false, bf[i], (short)0, acc[i], false, false);
    af = afn;
#pragma unroll
    for (int i = 0; i < 4; ++i) bf[i] = bfn[i];
  }
  // epilogue chunk
#pragma unroll
  for (int i = 0; i < 4; ++i)
    acc[i] = __builtin_amdgcn_wmma_f32_16x16x32_bf16(
        false, af, false, bf[i], (short)0, acc[i], false, false);

#pragma unroll
  for (int i = 0; i < 4; ++i) {
#pragma unroll
    for (int v = 0; v < 8; ++v) {
      const int row = m0 + v + ((lane >> 4) << 3);
      const int col = n0blk + i * 16 + (lane & 15);
      const float val = acc[i][v];
      if (mode == 3) {
        reinterpret_cast<float*>(out)[(size_t)row * D_MODEL + col] = val + bias[col];
      } else {
        const int b = row >> 10, l = row & 1023;
        const int h = col >> 6, dh = col & 63;
        const int bh = b * NHEADS + h;
        if (mode == 0)
          reinterpret_cast<__bf16*>(out)[((size_t)bh * SEQ + l) * DH + dh] = (__bf16)val;
        else if (mode == 1)
          reinterpret_cast<__bf16*>(out)[((size_t)bh * KVLEN + l) * DH + dh] = (__bf16)val;
        else  // mode 2: transposed V
          reinterpret_cast<__bf16*>(out)[((size_t)bh * DH + dh) * KVLEN + l] = (__bf16)val;
      }
    }
  }
}

// ---------------------------------------------------------------------------
// Flash attention.  grid = (16 qblocks, 80 heads), 128 threads = 4 waves;
// each wave owns a 16-row query tile, kv chunks of 32.
//   K_ext: [bh, kv, 64]  (dh-contiguous -> S B-frags are 2x b128)
//   Vt   : [bh, 64, kv]  (kv-contiguous -> PV B-frags are 2x b128)
// K frags pipelined across iterations; V frags issued before softmax.
// ---------------------------------------------------------------------------
__global__ void carc_attn_wmma(const __bf16* __restrict__ Q,
                               const __bf16* __restrict__ Kx,
                               const __bf16* __restrict__ Vt,
                               __bf16* __restrict__ Ctx) {
  __shared__ __bf16 lp[4 * 16 * 32];             // per-wave P tile (16x32)
  const int wave = threadIdx.x >> 5;
  const int lane = threadIdx.x & 31;
  const int bh = blockIdx.y;
  const int m0 = blockIdx.x * 64 + wave * 16;    // query row within head
  const int wb = wave * 16 * 32;                 // LDS base (halves)

  const __bf16* Qh = Q  + (size_t)bh * SEQ * DH;
  const __bf16* Kh = Kx + (size_t)bh * KVLEN * DH;
  const __bf16* Vh = Vt + (size_t)bh * DH * KVLEN;

  const v16bf qa0 = load_a_frag(Qh, DH, m0, 0);
  const v16bf qa1 = load_a_frag(Qh, DH, m0, 32);

  v8f acc[4] = {zero8(), zero8(), zero8(), zero8()};
  float mrow[8], lrow[8];
#pragma unroll
  for (int v = 0; v < 8; ++v) { mrow[v] = -1.0e30f; lrow[v] = 0.f; }

  // prologue: K fragments for kv0 = 0
  // kf[0]=(kv0,dh0..31) kf[1]=(kv0,dh32..63) kf[2]=(kv0+16,dh0..31) kf[3]=(kv0+16,dh32..63)
  v16bf kf[4];
  kf[0] = load_bT_frag(Kh, DH, 0, 0);
  kf[1] = load_bT_frag(Kh, DH, 0, 32);
  kf[2] = load_bT_frag(Kh, DH, 16, 0);
  kf[3] = load_bT_frag(Kh, DH, 16, 32);

  for (int kv0 = 0; kv0 < KVLEN; kv0 += 32) {
    // S tiles from the pipelined K fragments
    v8f s0 = zero8(), s1 = zero8();
    s0 = __builtin_amdgcn_wmma_f32_16x16x32_bf16(false, qa0, false, kf[0],
                                                 (short)0, s0, false, false);
    s0 = __builtin_amdgcn_wmma_f32_16x16x32_bf16(false, qa1, false, kf[1],
                                                 (short)0, s0, false, false);
    s1 = __builtin_amdgcn_wmma_f32_16x16x32_bf16(false, qa0, false, kf[2],
                                                 (short)0, s1, false, false);
    s1 = __builtin_amdgcn_wmma_f32_16x16x32_bf16(false, qa1, false, kf[3],
                                                 (short)0, s1, false, false);

    // issue next chunk's K frags and this chunk's V frags now, so their
    // latency hides behind the softmax VALU block below
    const int kvn = kv0 + 32;
    if (kvn < KVLEN) {
      kf[0] = load_bT_frag(Kh, DH, kvn, 0);
      kf[1] = load_bT_frag(Kh, DH, kvn, 32);
      kf[2] = load_bT_frag(Kh, DH, kvn + 16, 0);
      kf[3] = load_bT_frag(Kh, DH, kvn + 16, 32);
    }
    v16bf vb[4];
#pragma unroll
    for (int i = 0; i < 4; ++i)
      vb[i] = load_bT_frag(Vh, KVLEN, i * 16, kv0);

    // online softmax; row v lives across one 16-lane half (xor 1,2,4,8)
    float corr[8];
#pragma unroll
    for (int v = 0; v < 8; ++v) {
      const float a = s0[v] * SM_SCALE;
      const float b = s1[v] * SM_SCALE;
      float t = fmaxf(a, b);
      t = fmaxf(t, __shfl_xor(t, 1, 32));
      t = fmaxf(t, __shfl_xor(t, 2, 32));
      t = fmaxf(t, __shfl_xor(t, 4, 32));
      t = fmaxf(t, __shfl_xor(t, 8, 32));
      const float nm = fmaxf(mrow[v], t);
      const float c  = __expf(mrow[v] - nm);
      const float e0 = __expf(a - nm);
      const float e1 = __expf(b - nm);
      float rs = e0 + e1;
      rs += __shfl_xor(rs, 1, 32);
      rs += __shfl_xor(rs, 2, 32);
      rs += __shfl_xor(rs, 4, 32);
      rs += __shfl_xor(rs, 8, 32);
      lrow[v] = lrow[v] * c + rs;
      mrow[v] = nm;
      corr[v] = c;
      // park P (bf16) in LDS: row = v + 8*(lane>=16), cols lane&15 / +16
      const int r  = v + ((lane >> 4) << 3);
      const int cI = lane & 15;
      lp[wb + r * 32 + cI]      = (__bf16)e0;
      lp[wb + r * 32 + 16 + cI] = (__bf16)e1;
    }

#pragma unroll
    for (int i = 0; i < 4; ++i)
#pragma unroll
      for (int v = 0; v < 8; ++v) acc[i][v] *= corr[v];

    // same-wave LDS RAW across lanes: fence the DS stores explicitly
    asm volatile("s_wait_dscnt 0x0" ::: "memory");

    // reload P as an A fragment: two 16B ds loads per lane
    v16bf pa;
    {
      const int m  = lane & 15;
      const int kb = (lane >> 4) << 3;
      const __bf16* p = &lp[wb + m * 32 + kb];
      union U { uint4 u[2]; v16bf v; } x;
      x.u[0] = *reinterpret_cast<const uint4*>(p);
      x.u[1] = *reinterpret_cast<const uint4*>(p + 16);
      pa = x.v;
    }
#pragma unroll
    for (int i = 0; i < 4; ++i)
      acc[i] = __builtin_amdgcn_wmma_f32_16x16x32_bf16(
          false, pa, false, vb[i], (short)0, acc[i], false, false);
  }

  // normalize + store context in batch_to_head layout [4096, 1280] bf16
  const int b = bh / NHEADS, h = bh % NHEADS;
#pragma unroll
  for (int i = 0; i < 4; ++i) {
#pragma unroll
    for (int v = 0; v < 8; ++v) {
      const int l   = m0 + v + ((lane >> 4) << 3);
      const int col = h * DH + i * 16 + (lane & 15);
      const float val = acc[i][v] / lrow[v];
      Ctx[((size_t)b * SEQ + l) * D_MODEL + col] = (__bf16)val;
    }
  }
}

// ---------------------------------------------------------------------------
// Host launcher
// ---------------------------------------------------------------------------
extern "C" void kernel_launch(void* const* d_in, const int* in_sizes, int n_in,
                              void* d_out, int out_size, void* d_ws, size_t ws_size,
                              hipStream_t stream) {
  (void)in_sizes; (void)n_in; (void)out_size; (void)ws_size;

  const float* hs   = (const float*)d_in[0];
  const float* Wq   = (const float*)d_in[1];
  const float* Wk   = (const float*)d_in[2];
  const float* Wv   = (const float*)d_in[3];
  const float* Wo   = (const float*)d_in[4];
  const float* bo   = (const float*)d_in[5];
  const float* K_bg = (const float*)d_in[6];
  const float* V_bg = (const float*)d_in[7];

  const size_t nHS = (size_t)MROWS * D_MODEL;        // 5,242,880
  const size_t nW  = (size_t)D_MODEL * D_MODEL;      // 1,638,400
  const size_t nQ  = (size_t)BH * SEQ * DH;          // 5,242,880
  const size_t nKX = (size_t)BH * KVLEN * DH;        // 10,485,760

  __bf16* HSb  = (__bf16*)d_ws;
  __bf16* WqT  = HSb  + nHS;
  __bf16* WkT  = WqT  + nW;
  __bf16* WvT  = WkT  + nW;
  __bf16* WoT  = WvT  + nW;
  __bf16* Qb   = WoT  + nW;
  __bf16* Kext = Qb   + nQ;
  __bf16* VtB  = Kext + nKX;
  __bf16* Ctxb = VtB  + nKX;

  const int TB = 256;
  carc_cvt_bf16<<<(int)((nHS + TB - 1) / TB), TB, 0, stream>>>(hs, HSb, (int)nHS);
  carc_cvt_w_t <<<(int)((nW + TB - 1) / TB), TB, 0, stream>>>(Wq, WqT, (int)nW);
  carc_cvt_w_t <<<(int)((nW + TB - 1) / TB), TB, 0, stream>>>(Wk, WkT, (int)nW);
  carc_cvt_w_t <<<(int)((nW + TB - 1) / TB), TB, 0, stream>>>(Wv, WvT, (int)nW);
  carc_cvt_w_t <<<(int)((nW + TB - 1) / TB), TB, 0, stream>>>(Wo, WoT, (int)nW);
  carc_cvt_bg_k <<<(int)((nQ + TB - 1) / TB), TB, 0, stream>>>(K_bg, Kext, (int)nQ);
  carc_cvt_bg_vT<<<(int)((nQ + TB - 1) / TB), TB, 0, stream>>>(V_bg, VtB, (int)nQ);

  dim3 ggrid(D_MODEL / 64, MROWS / 64);              // (20, 64)
  carc_gemm_wmma<<<ggrid, 128, 0, stream>>>(HSb, WqT, 0, nullptr, Qb);
  carc_gemm_wmma<<<ggrid, 128, 0, stream>>>(HSb, WkT, 1, nullptr, Kext);
  carc_gemm_wmma<<<ggrid, 128, 0, stream>>>(HSb, WvT, 2, nullptr, VtB);

  dim3 agrid(SEQ / 64, BH);                          // (16, 80)
  carc_attn_wmma<<<agrid, 128, 0, stream>>>(Qb, Kext, VtB, Ctxb);

  carc_gemm_wmma<<<ggrid, 128, 0, stream>>>(Ctxb, WoT, 3, bo, (float*)d_out);
}
